// InjectedEncoder_87445534147108
// MI455X (gfx1250) — matrix operand order repaired and checked by reading, not compiled, not run on hardware
//
#include <hip/hip_runtime.h>
#include <hip/hip_bf16.h>

// ---------------------------------------------------------------------------
// InjectedEncoder forward for gfx1250 (MI455X).  bf16 WMMA for channel GEMMs,
// fragment-order (swizzled) LDS staging so every WMMA operand is loaded as
// 2 x ds_load_b128, async global->LDS staging of activation tiles.
// ---------------------------------------------------------------------------

#define IE_ASYNC 1   // use global_load_async_to_lds_b128 for A-tile staging

typedef __attribute__((ext_vector_type(16))) __bf16 v16bf;
typedef __attribute__((ext_vector_type(8)))  float  v8f;

#define NF    64
#define FIN   192
#define FH    128
#define FOUT  64
#define PTOT  61824
// dyn param offsets (elements, per sample)
#define O_WIN  0
#define O_BIN  24576
#define O_WMID 24704
#define O_BMID 41088
#define O_WOUT 41216
#define O_BOUT 49408
#define O_WSK  49472
#define O_BSK  61760

static __device__ __forceinline__ float lrelu(float v) { return v > 0.0f ? v : 0.2f * v; }

static __device__ __forceinline__ unsigned short f2bf(float f) {
  unsigned int u = __float_as_uint(f);
  u += 0x7fffu + ((u >> 16) & 1u);           // round to nearest even
  return (unsigned short)(u >> 16);
}

// ---- fragment-order ("swizzled") layouts --------------------------------
// A 16x32 bf16 block: lane = row + 16*(koff>=16),
//   elem e = koff&15.  One block = 32 lanes * 16 elems = 512 us.
static __device__ __forceinline__ int swzA(int koff, int row) {
  int lane = row + (koff & 16);
  int e = koff & 15;
  return lane * 16 + e;
}
// B 32x16 bf16 block: lane = col + 16*(koff>=16), elem e = koff&15.
static __device__ __forceinline__ int swzB(int koff, int col) {
  int lane = (col & 15) + (koff & 16);
  int e = koff & 15;
  return lane * 16 + e;
}
// load one fragment (this lane's 32 contiguous bytes) from a swizzled block
static __device__ __forceinline__ v16bf frag_ld(const unsigned short* blk) {
  return *(const v16bf*)(blk + (threadIdx.x & 31) * 16);
}

// ---------------------------------------------------------------------------
// 0) transpose + bf16 quantize hyp_w:  wT[n][k] = bf16(hw[k][n])
// ---------------------------------------------------------------------------
__global__ void ie_hyp_tr(const float* __restrict__ hw, unsigned short* __restrict__ wT) {
  size_t idx = (size_t)blockIdx.x * blockDim.x + threadIdx.x;
  if (idx >= (size_t)512 * PTOT) return;
  int k = (int)(idx / PTOT), n = (int)(idx % PTOT);
  wT[(size_t)n * 512 + k] = f2bf(hw[idx]);
}

// ---------------------------------------------------------------------------
// 1) Hypernetwork GEMM: dyn[8,61824] = inj[8,512] @ hyp_w[512,61824] + hyp_b
//    M padded 8->16; one wave per 16-column tile; K loop of 16 WMMAs.
// ---------------------------------------------------------------------------
__global__ void ie_hyper_kernel(const float* __restrict__ inj,
                                const float* __restrict__ hw,
                                const unsigned short* __restrict__ wT, int useT,
                                const float* __restrict__ hb,
                                float* __restrict__ dynf,
                                unsigned short* __restrict__ dynb) {
  __shared__ __align__(32) unsigned short la[16 * 512];   // A in fragment order
  const int tid = threadIdx.x;
  for (int i = tid; i < 16 * 512; i += 256) {
    int m = i >> 9, k = i & 511;
    unsigned short v = (m < 8) ? f2bf(inj[m * 512 + k]) : (unsigned short)0;
    la[(k >> 5) * 512 + swzA(k & 31, m)] = v;
  }
  __syncthreads();

  const int wave = tid >> 5, lane = tid & 31;
  const int nt = blockIdx.x * 8 + wave;       // 0..3863
  const int n  = nt * 16 + (lane & 15);
  const int khiB = (lane & 16) ? 16 : 0;

  v8f acc = {};
  for (int kb = 0; kb < 16; ++kb) {
    v16bf a = frag_ld(la + kb * 512);
    v16bf bw;
    if (useT) {
      bw = *(const v16bf*)(wT + (size_t)n * 512 + kb * 32 + khiB);
    } else {
#pragma unroll
      for (int e = 0; e < 16; ++e) {
        union { unsigned short u; __bf16 h; } c;
        c.u = f2bf(hw[(size_t)(kb * 32 + khiB + e) * PTOT + n]);
        bw[e] = c.h;
      }
    }
    acc = __builtin_amdgcn_wmma_f32_16x16x32_bf16(false, a, false, bw, (short)0, acc,
                                                  false, false);
  }
  const float bias = hb[n];
  if (lane < 16) {                            // rows 0..7 = batch (8..15 are pad)
#pragma unroll
    for (int r = 0; r < 8; ++r) {
      float v = acc[r] + bias;
      dynf[(size_t)r * PTOT + n] = v;
      dynb[(size_t)r * PTOT + n] = f2bf(v);
    }
  }
}

// ---------------------------------------------------------------------------
// 2) Input 1x1 conv: emb0 = in_w[64,3] * x + in_b
// ---------------------------------------------------------------------------
__global__ void ie_inconv_kernel(const float* __restrict__ x,
                                 const float* __restrict__ in_w,
                                 const float* __restrict__ in_b,
                                 float* __restrict__ emb0) {
  __shared__ float w[192];
  __shared__ float bb[64];
  const int tid = threadIdx.x;
  if (tid < 192) w[tid] = in_w[tid];
  if (tid < 64)  bb[tid] = in_b[tid];
  __syncthreads();
  size_t idx = (size_t)blockIdx.x * blockDim.x + tid;      // (b,o,pix)
  const size_t npix = 16384;
  if (idx >= (size_t)8 * 64 * npix) return;
  size_t pix = idx % npix;
  int bo = (int)(idx / npix);
  int b = bo >> 6, o = bo & 63;
  const float* xb = x + (size_t)b * 3 * npix + pix;
  float v = bb[o] + w[o * 3 + 0] * xb[0] + w[o * 3 + 1] * xb[npix] + w[o * 3 + 2] * xb[2 * npix];
  emb0[idx] = v;
}

// ---------------------------------------------------------------------------
// 3) input res_block via WMMA, in place on emb0:
//    out = v + 0.1*conv(act(conv(act(v),w0)+b0 -> act), w1)+b1
//    grid (chunks, B); one 16-pixel tile per wave; K=64 -> 2 A-blocks.
// ---------------------------------------------------------------------------
__global__ void ie_inres_kernel(float* __restrict__ emb,
                                const float* __restrict__ w0g, const float* __restrict__ b0g,
                                const float* __restrict__ w1g, const float* __restrict__ b1g) {
  extern __shared__ unsigned char smem_raw[];
  unsigned short* lw0 = (unsigned short*)smem_raw;        // 2kb x 4nt blocks
  unsigned short* lw1 = lw0 + 4096;                       // 2kb x 4nt blocks
  unsigned short* tiles = lw1 + 4096;                     // 8 waves * (1024+1024)
  float* lb = (float*)(tiles + 8 * 2048);                 // 128 f32

  const int tid  = threadIdx.x;
  const int wave = tid >> 5;
  const int lane = tid & 31;
  const int b    = blockIdx.y;
  const size_t npix = 16384;
  const int ntiles = 1024;

  for (int i = tid; i < 4096; i += 256) {                 // src [o=64][k=64]
    int o = i >> 6, k = i & 63;
    int dst = ((k >> 5) * 4 + (o >> 4)) * 512 + swzB(k & 31, o);
    lw0[dst] = f2bf(w0g[i]);
    lw1[dst] = f2bf(w1g[i]);
  }
  if (tid < 64) { lb[tid] = b0g[tid]; lb[64 + tid] = b1g[tid]; }
  __syncthreads();

  unsigned short* a0 = tiles + wave * 2048;               // 2 A-blocks (act(v))
  unsigned short* hh = a0 + 1024;                         // 2 A-blocks

  const int colL   = lane & 15;
  const int rowadd = (lane & 16) ? 8 : 0;
  const int iters  = (ntiles + (int)gridDim.x * 8 - 1) / ((int)gridDim.x * 8);

  for (int it = 0; it < iters; ++it) {
    const int t = (it * (int)gridDim.x + (int)blockIdx.x) * 8 + wave;
    const bool active = (t < ntiles);                     // wave-uniform

    if (active) {                                         // stage act(v) tile
      for (int c = lane; c < 64; c += 32) {
        const float* vp = emb + ((size_t)b * 64 + c) * npix + (size_t)t * 16;
        unsigned short* d = a0 + (c >> 5) * 512;
        const int koff = c & 31;
#pragma unroll
        for (int px = 0; px < 16; ++px) d[swzA(koff, px)] = f2bf(lrelu(vp[px]));
      }
    }
    __syncthreads();

    v16bf aT[2];
#pragma unroll
    for (int kb = 0; kb < 2; ++kb) aT[kb] = frag_ld(a0 + kb * 512);

    // GEMM1: hh = act(act(v) x w0 + b0)
    for (int nt = 0; nt < 4; ++nt) {
      v8f acc = {};
#pragma unroll
      for (int kb = 0; kb < 2; ++kb)
        acc = __builtin_amdgcn_wmma_f32_16x16x32_bf16(
            false, aT[kb], false, frag_ld(lw0 + (kb * 4 + nt) * 512), (short)0, acc,
            false, false);
      const int col = nt * 16 + colL;
      const float bias = lb[col];
      unsigned short* d = hh + (col >> 5) * 512;
      const int koff = col & 31;
#pragma unroll
      for (int r = 0; r < 8; ++r)
        d[swzA(koff, r + rowadd)] = f2bf(lrelu(acc[r] + bias));
    }
    __syncthreads();

    v16bf hT[2];
#pragma unroll
    for (int kb = 0; kb < 2; ++kb) hT[kb] = frag_ld(hh + kb * 512);

    // GEMM2 + residual: emb = v + 0.1*(hh x w1 + b1)    (in place, tile-exclusive)
    for (int nt = 0; nt < 4; ++nt) {
      v8f acc = {};
#pragma unroll
      for (int kb = 0; kb < 2; ++kb)
        acc = __builtin_amdgcn_wmma_f32_16x16x32_bf16(
            false, hT[kb], false, frag_ld(lw1 + (kb * 4 + nt) * 512), (short)0, acc,
            false, false);
      if (active) {
        const int col = nt * 16 + colL;
        const float bias = lb[64 + col];
        float* op = emb + ((size_t)b * 64 + col) * npix + (size_t)t * 16 + rowadd;
#pragma unroll
        for (int r = 0; r < 8; ++r) op[r] = op[r] + 0.1f * (acc[r] + bias);
      }
    }
    __syncthreads();
  }
}

// ---------------------------------------------------------------------------
// 4) sin_sobel: writes ss in *tile-fragment order*:
//    ss[((b*ntiles + t)*6 + kb)*512 + swzA(koff, row)],  k=channel, row=pix&15.
//    Each 16-pixel tile is 6 KB contiguous -> async-copyable to LDS.
// ---------------------------------------------------------------------------
__global__ void ie_sinsobel_kernel(const float* __restrict__ emb,
                                   unsigned short* __restrict__ ss, int H, int W) {
  size_t idx = (size_t)blockIdx.x * blockDim.x + threadIdx.x;
  const size_t npix = (size_t)H * W;
  if (idx >= (size_t)8 * 64 * npix) return;
  size_t pix = idx % npix;
  int bc = (int)(idx / npix);
  int y = (int)(pix / W), x = (int)(pix % W);
  const float* p = emb + (size_t)bc * npix;
  float v[3][3];
#pragma unroll
  for (int dy = 0; dy < 3; ++dy)
#pragma unroll
    for (int dx = 0; dx < 3; ++dx) {
      int yy = y + dy - 1, xx = x + dx - 1;
      v[dy][dx] = (yy >= 0 && yy < H && xx >= 0 && xx < W) ? p[(size_t)yy * W + xx] : 0.0f;
    }
  float c  = v[1][1];
  float sx = (v[0][0] - v[0][2] + 2.0f * (v[1][0] - v[1][2]) + v[2][0] - v[2][2]) * 0.125f;
  float sy = (v[0][0] + 2.0f * v[0][1] + v[0][2] - v[2][0] - 2.0f * v[2][1] - v[2][2]) * 0.125f;
  int b = bc >> 6, ch = bc & 63;
  int ntiles = (int)(npix >> 4);
  int t = (int)(pix >> 4), row = (int)(pix & 15);
  unsigned short* tb = ss + (((size_t)b * ntiles + t) * 6) * 512;
  tb[((ch)        >> 5) * 512 + swzA(ch & 31, row)]          = f2bf(c);
  tb[((ch + 64)   >> 5) * 512 + swzA((ch + 64) & 31, row)]   = f2bf(sx);
  tb[((ch + 128)  >> 5) * 512 + swzA((ch + 128) & 31, row)]  = f2bf(sy);
}

// ---------------------------------------------------------------------------
// 5) dyna residual block: WMMA chain (skip + in->mid->out) per 16-pixel tile.
//    grid = (chunks, B); 256 threads = 8 waves; dynamic LDS ~234 KB.
// ---------------------------------------------------------------------------
__global__ void ie_dyna_kernel(const unsigned short* __restrict__ ss,
                               const unsigned short* __restrict__ dynb,
                               const float* __restrict__ dynf,
                               float* __restrict__ out_new, int H, int W) {
  extern __shared__ unsigned char smem_raw[];
  unsigned short* lw_in  = (unsigned short*)smem_raw;   // 6kb x 8nt blocks
  unsigned short* lw_mid = lw_in + FIN * FH;            // 4kb x 8nt
  unsigned short* lw_out = lw_mid + FH * FH;            // 4kb x 4nt
  unsigned short* lw_sk  = lw_out + FH * FOUT;          // 6kb x 4nt
  unsigned short* tiles  = lw_sk + FIN * FOUT;          // 8 waves * 7168
  float* lbias = (float*)(tiles + 8 * (FIN * 16 + 2 * FH * 16));  // 384 f32

  const int tid  = threadIdx.x;
  const int wave = tid >> 5;
  const int lane = tid & 31;
  const int b    = blockIdx.y;
  const size_t npix = (size_t)H * W;
  const int ntiles  = (int)(npix >> 4);

  // stage per-sample weights into LDS in B-fragment order
  const unsigned short* dp  = dynb + (size_t)b * PTOT;
  const float*          dpf = dynf + (size_t)b * PTOT;
  for (int i = tid; i < FH * FIN; i += 256) {           // src [o=128][k=192]
    int o = i / FIN, k = i % FIN;
    lw_in[((k >> 5) * 8 + (o >> 4)) * 512 + swzB(k & 31, o)] = dp[O_WIN + i];
  }
  for (int i = tid; i < FH * FH; i += 256) {            // src [o=128][k=128]
    int o = i >> 7, k = i & 127;
    lw_mid[((k >> 5) * 8 + (o >> 4)) * 512 + swzB(k & 31, o)] = dp[O_WMID + i];
  }
  for (int i = tid; i < FOUT * FH; i += 256) {          // src [o=64][k=128]
    int o = i >> 7, k = i & 127;
    lw_out[((k >> 5) * 4 + (o >> 4)) * 512 + swzB(k & 31, o)] = dp[O_WOUT + i];
  }
  for (int i = tid; i < FOUT * FIN; i += 256) {         // src [o=64][k=192]
    int o = i / FIN, k = i % FIN;
    lw_sk[((k >> 5) * 4 + (o >> 4)) * 512 + swzB(k & 31, o)] = dp[O_WSK + i];
  }
  for (int i = tid; i < FH; i += 256) { lbias[i] = dpf[O_BIN + i]; lbias[FH + i] = dpf[O_BMID + i]; }
  for (int i = tid; i < FOUT; i += 256) { lbias[256 + i] = dpf[O_BOUT + i]; lbias[320 + i] = dpf[O_BSK + i]; }
  __syncthreads();

  unsigned short* a_tile = tiles + wave * (FIN * 16 + 2 * FH * 16);  // 6 A-blocks
  unsigned short* h1     = a_tile + FIN * 16;                        // 4 A-blocks
  unsigned short* h2     = h1 + FH * 16;                             // 4 A-blocks
  const unsigned a_tile_loff = (unsigned)((char*)a_tile - (char*)smem_raw);

  const int colL   = lane & 15;
  const int rowadd = (lane & 16) ? 8 : 0;
  const int iters  = (ntiles + (int)gridDim.x * 8 - 1) / ((int)gridDim.x * 8);

  for (int it = 0; it < iters; ++it) {
    const int t = (it * (int)gridDim.x + (int)blockIdx.x) * 8 + wave;
    const bool active = (t < ntiles);                     // wave-uniform

    if (active) {                                         // stage 6KB tile -> LDS
      const unsigned short* sp = ss + ((size_t)b * ntiles + t) * (6 * 512);
#if IE_ASYNC
#pragma unroll
      for (int j = 0; j < 12; ++j) {                      // 12 x (32 lanes x 16B)
        int i = j * 32 + lane;
        unsigned long long ga = (unsigned long long)(const void*)(sp + i * 8);
        unsigned lo = a_tile_loff + i * 16;
        asm volatile("global_load_async_to_lds_b128 %0, %1, off"
                     :: "v"(lo), "v"(ga) : "memory");
      }
      asm volatile("s_wait_asynccnt 0x0" ::: "memory");
#else
#pragma unroll
      for (int j = 0; j < 12; ++j) {
        int i = j * 32 + lane;
        ((uint4*)a_tile)[i] = ((const uint4*)sp)[i];
      }
#endif
    }
    __syncthreads();

    v16bf aT[6];
#pragma unroll
    for (int kb = 0; kb < 6; ++kb) aT[kb] = frag_ld(a_tile + kb * 512);

    // GEMM1: h1 = act(A0[16x192] x Win[192x128] + b_in)
    for (int nt = 0; nt < 8; ++nt) {
      v8f acc = {};
#pragma unroll
      for (int kb = 0; kb < 6; ++kb)
        acc = __builtin_amdgcn_wmma_f32_16x16x32_bf16(
            false, aT[kb], false, frag_ld(lw_in + (kb * 8 + nt) * 512), (short)0, acc,
            false, false);
      const int col = nt * 16 + colL;
      const float bias = lbias[col];
      unsigned short* d = h1 + (col >> 5) * 512;
      const int koff = col & 31;
#pragma unroll
      for (int r = 0; r < 8; ++r)
        d[swzA(koff, r + rowadd)] = f2bf(lrelu(acc[r] + bias));
    }
    __syncthreads();

    v16bf hT[4];
#pragma unroll
    for (int kb = 0; kb < 4; ++kb) hT[kb] = frag_ld(h1 + kb * 512);

    // GEMM2: h2 = act(h1[16x128] x Wmid[128x128] + b_mid)
    for (int nt = 0; nt < 8; ++nt) {
      v8f acc = {};
#pragma unroll
      for (int kb = 0; kb < 4; ++kb)
        acc = __builtin_amdgcn_wmma_f32_16x16x32_bf16(
            false, hT[kb], false, frag_ld(lw_mid + (kb * 8 + nt) * 512), (short)0, acc,
            false, false);
      const int col = nt * 16 + colL;
      const float bias = lbias[FH + col];
      unsigned short* d = h2 + (col >> 5) * 512;
      const int koff = col & 31;
#pragma unroll
      for (int r = 0; r < 8; ++r)
        d[swzA(koff, r + rowadd)] = f2bf(lrelu(acc[r] + bias));
    }
    __syncthreads();

    v16bf h2T[4];
#pragma unroll
    for (int kb = 0; kb < 4; ++kb) h2T[kb] = frag_ld(h2 + kb * 512);

    // GEMM3+4 fused via C operand: out = A0 x Wsk  +  h2 x Wout  (+ biases)
    for (int nt = 0; nt < 4; ++nt) {
      v8f acc = {};
#pragma unroll
      for (int kb = 0; kb < 6; ++kb)
        acc = __builtin_amdgcn_wmma_f32_16x16x32_bf16(
            false, aT[kb], false, frag_ld(lw_sk + (kb * 4 + nt) * 512), (short)0, acc,
            false, false);
#pragma unroll
      for (int kb = 0; kb < 4; ++kb)
        acc = __builtin_amdgcn_wmma_f32_16x16x32_bf16(
            false, h2T[kb], false, frag_ld(lw_out + (kb * 4 + nt) * 512), (short)0, acc,
            false, false);
      if (active) {
        const int col = nt * 16 + colL;
        const float bias = lbias[256 + col] + lbias[320 + col];
        float* op = out_new + ((size_t)b * 64 + col) * npix + (size_t)t * 16 + rowadd;
#pragma unroll
        for (int r = 0; r < 8; ++r) op[r] = acc[r] + bias;
      }
    }
    __syncthreads();
  }
}

// ---------------------------------------------------------------------------
// 6) combine + gauss(3x3,zero-pad) + 2x2 mean, fused as a 4x4 tap filter on
//    t = emb + clip(leak)*out_new.  Writes next pyramid level.
// ---------------------------------------------------------------------------
__global__ void ie_combine_down_kernel(const float* __restrict__ emb,
                                       const float* __restrict__ onew,
                                       const float* __restrict__ leakp,
                                       float* __restrict__ embo, int H, int W) {
  const int Ho = H >> 1, Wo = W >> 1;
  const size_t npix = (size_t)H * W, nop = (size_t)Ho * Wo;
  size_t idx = (size_t)blockIdx.x * blockDim.x + threadIdx.x;
  if (idx >= (size_t)8 * 64 * nop) return;
  size_t opix = idx % nop;
  int bc = (int)(idx / nop);
  int oy = (int)(opix / Wo), ox = (int)(opix % Wo);
  float lk = leakp[0];
  lk = fminf(fmaxf(lk, 0.001f), 1000.0f);
  const float* pe = emb  + (size_t)bc * npix;
  const float* pn = onew + (size_t)bc * npix;
  const float GN0 = 0.27406862f, GN1 = 0.45186276f;
  const float wv[4] = { 0.5f * GN0, 0.5f * (GN0 + GN1), 0.5f * (GN0 + GN1), 0.5f * GN0 };
  float s = 0.0f;
#pragma unroll
  for (int iy = 0; iy < 4; ++iy) {
    int yy = 2 * oy - 1 + iy;
    if (yy < 0 || yy >= H) continue;
#pragma unroll
    for (int ix = 0; ix < 4; ++ix) {
      int xx = 2 * ox - 1 + ix;
      if (xx < 0 || xx >= W) continue;
      float t = pe[(size_t)yy * W + xx] + lk * pn[(size_t)yy * W + xx];
      s += wv[iy] * wv[ix] * t;
    }
  }
  embo[(size_t)bc * nop + opix] = s;
}

// ---------------------------------------------------------------------------
// 7) tail head: outrb res_block on emb6 [8,64,2,2] + spatial mean -> h0[8,64]
// ---------------------------------------------------------------------------
__global__ void ie_tail_head_kernel(const float* __restrict__ emb6,
                                    const float* __restrict__ w0, const float* __restrict__ b0,
                                    const float* __restrict__ w1, const float* __restrict__ b1,
                                    float* __restrict__ h0out) {
  __shared__ float tact[8 * 64 * 4];
  __shared__ float sb0[64], sb1[64];
  if (threadIdx.x < 64) { sb0[threadIdx.x] = b0[threadIdx.x]; sb1[threadIdx.x] = b1[threadIdx.x]; }
  __syncthreads();
  for (int it = threadIdx.x; it < 2048; it += blockDim.x) {   // (b,i,p)
    int p = it & 3, i = (it >> 2) & 63, b = it >> 8;
    const float* base = emb6 + (size_t)b * 64 * 4 + p;
    float s = sb0[i];
    for (int j = 0; j < 64; ++j) s += w0[i * 64 + j] * lrelu(base[j * 4]);
    tact[(b * 64 + i) * 4 + p] = lrelu(s);
  }
  __syncthreads();
  for (int it = threadIdx.x; it < 512; it += blockDim.x) {    // (b,o)
    int o = it & 63, b = it >> 6;
    float acc = 0.0f;
    for (int p = 0; p < 4; ++p) {
      float s = sb1[o];
      for (int i = 0; i < 64; ++i) s += w1[o * 64 + i] * tact[(b * 64 + i) * 4 + p];
      acc += emb6[((size_t)b * 64 + o) * 4 + p] + 0.1f * s;
    }
    h0out[b * 64 + o] = acc * 0.25f;
  }
}

// ---------------------------------------------------------------------------
// 8) tail GEMMs (8-row)
// ---------------------------------------------------------------------------
__global__ void ie_tg_act(const float* __restrict__ X, int K,
                          const float* __restrict__ W, const float* __restrict__ Bv,
                          float* __restrict__ Y, int N, int preact, int actout) {
  int idx = blockIdx.x * blockDim.x + threadIdx.x;
  if (idx >= 8 * N) return;
  int n = idx % N, b = idx / N;
  const float* xr = X + (size_t)b * K;
  float s = Bv[n];
  for (int k = 0; k < K; ++k) {
    float a = xr[k];
    if (preact) a = lrelu(a);
    s += a * W[(size_t)k * N + n];
  }
  if (actout) s = lrelu(s);
  Y[idx] = s;
}

__global__ void ie_tg_axpy(const float* __restrict__ X, int K,
                           const float* __restrict__ W, const float* __restrict__ Bv,
                           const float* __restrict__ Skip, float* __restrict__ Y, int N) {
  int idx = blockIdx.x * blockDim.x + threadIdx.x;
  if (idx >= 8 * N) return;
  int n = idx % N, b = idx / N;
  const float* xr = X + (size_t)b * K;
  float s = Bv[n];
  for (int k = 0; k < K; ++k) s += xr[k] * W[(size_t)k * N + n];
  Y[idx] = Skip[idx] + 0.1f * s;
}

// ---------------------------------------------------------------------------
// host side
// ---------------------------------------------------------------------------
extern "C" void kernel_launch(void* const* d_in, const int* in_sizes, int n_in,
                              void* d_out, int out_size, void* d_ws, size_t ws_size,
                              hipStream_t stream) {
  const float* x      = (const float*)d_in[0];
  const float* inj    = (const float*)d_in[1];
  const float* leak   = (const float*)d_in[2];
  const float* in_w   = (const float*)d_in[3];
  const float* in_b   = (const float*)d_in[4];
  const float* inrb_w0 = (const float*)d_in[5];
  const float* inrb_b0 = (const float*)d_in[6];
  const float* inrb_w1 = (const float*)d_in[7];
  const float* inrb_b1 = (const float*)d_in[8];
  const float* hyp_w  = (const float*)d_in[9];
  const float* hyp_b  = (const float*)d_in[10];
  const float* outrb_w0 = (const float*)d_in[11];
  const float* outrb_b0 = (const float*)d_in[12];
  const float* outrb_w1 = (const float*)d_in[13];
  const float* outrb_b1 = (const float*)d_in[14];
  const float* l1_sw = (const float*)d_in[15];
  const float* l1_sb = (const float*)d_in[16];
  const float* l1_w1 = (const float*)d_in[17];
  const float* l1_b1 = (const float*)d_in[18];
  const float* l1_w2 = (const float*)d_in[19];
  const float* l1_b2 = (const float*)d_in[20];
  const float* l2_w1 = (const float*)d_in[21];
  const float* l2_b1 = (const float*)d_in[22];
  const float* l2_w2 = (const float*)d_in[23];
  const float* l2_b2 = (const float*)d_in[24];
  const float* lo_w  = (const float*)d_in[25];
  const float* lo_b  = (const float*)d_in[26];
  float* dout = (float*)d_out;

  // d_out layout: lat[4096], emb0..emb6
  static const size_t embOff[8] = {4096ull, 8392704ull, 10489856ull, 11014144ull,
                                   11145216ull, 11177984ull, 11186176ull, 11188224ull};

  // workspace partition (256B aligned)
  size_t cur = 0;
  auto take = [&](size_t bytes) -> void* {
    void* p = (char*)d_ws + cur;
    cur += bytes;
    cur = (cur + 255) & ~(size_t)255;
    return p;
  };
  float*          dynf = (float*)take((size_t)8 * PTOT * 4);
  unsigned short* dynb = (unsigned short*)take((size_t)8 * PTOT * 2);
  unsigned short* ssb  = (unsigned short*)take((size_t)8 * 192 * 16384 * 2);
  float*          onew = (float*)take((size_t)8 * 64 * 16384 * 4);
  float* h0  = (float*)take(8 * 64 * 4);
  float* a1  = (float*)take(8 * 1024 * 4);
  float* xs  = (float*)take(8 * 512 * 4);
  float* h1v = (float*)take(8 * 512 * 4);
  float* bbv = (float*)take(8 * 512 * 4);
  float* h2v = (float*)take(8 * 512 * 4);
  // optional transposed bf16 hyp_w (63 MB) -- only if workspace allows
  const size_t wTbytes = (size_t)PTOT * 512 * 2;
  int useT = (cur + wTbytes <= ws_size) ? 1 : 0;
  unsigned short* wT = useT ? (unsigned short*)take(wTbytes) : (unsigned short*)0;

  // 1) hypernetwork GEMM (WMMA)
  if (useT) {
    int trBlocks = (int)(((size_t)512 * PTOT + 255) / 256);
    ie_hyp_tr<<<trBlocks, 256, 0, stream>>>(hyp_w, wT);
  }
  ie_hyper_kernel<<<483, 256, 0, stream>>>(inj, hyp_w, wT, useT, hyp_b, dynf, dynb);

  // 2) input conv + input res_block (WMMA) -> emb0
  float* emb0 = dout + embOff[0];
  const int bigBlocks = (int)(((size_t)8 * 64 * 16384 + 255) / 256);
  ie_inconv_kernel<<<bigBlocks, 256, 0, stream>>>(x, in_w, in_b, emb0);
  const size_t inresSmem = (size_t)(4096 + 4096) * 2 + (size_t)8 * 2048 * 2 + 128 * 4;
  ie_inres_kernel<<<dim3(32, 8), 256, inresSmem, stream>>>(emb0, inrb_w0, inrb_b0,
                                                           inrb_w1, inrb_b1);

  // 3) pyramid levels
  const size_t dynaSmem =
      (size_t)(FIN * FH + FH * FH + FH * FOUT + FIN * FOUT) * 2   // weights (bf16)
      + (size_t)8 * (FIN * 16 + 2 * FH * 16) * 2                  // per-wave tiles
      + 384 * 4;                                                   // biases
  for (int lvl = 0; lvl < 6; ++lvl) {
    const int H = 128 >> lvl, W = H;
    const size_t npix = (size_t)H * W;
    const int ntiles = (int)(npix >> 4);
    const float* embI = dout + embOff[lvl];
    float* embO = dout + embOff[lvl + 1];

    int ssBlocks = (int)(((size_t)8 * 64 * npix + 255) / 256);
    ie_sinsobel_kernel<<<ssBlocks, 256, 0, stream>>>(embI, ssb, H, W);

    int gX = ntiles / 8;
    if (gX < 1) gX = 1;
    if (gX > 32) gX = 32;
    ie_dyna_kernel<<<dim3(gX, 8), 256, dynaSmem, stream>>>(ssb, dynb, dynf, onew, H, W);

    int cbBlocks = (int)(((size_t)8 * 64 * (npix >> 2) + 255) / 256);
    ie_combine_down_kernel<<<cbBlocks, 256, 0, stream>>>(embI, onew, leak, embO, H, W);
  }

  // 4) tail: outrb res_block + mean, then MLP stack -> lat
  const float* emb6 = dout + embOff[6];
  ie_tail_head_kernel<<<1, 256, 0, stream>>>(emb6, outrb_w0, outrb_b0, outrb_w1, outrb_b1, h0);
  ie_tg_act <<<(8 * 1024 + 255) / 256, 256, 0, stream>>>(h0, 64, l1_w1, l1_b1, a1, 1024, 1, 1);
  ie_tg_act <<<(8 * 512 + 255) / 256, 256, 0, stream>>>(h0, 64, l1_sw, l1_sb, xs, 512, 0, 0);
  ie_tg_axpy<<<(8 * 512 + 255) / 256, 256, 0, stream>>>(a1, 1024, l1_w2, l1_b2, xs, h1v, 512);
  ie_tg_act <<<(8 * 512 + 255) / 256, 256, 0, stream>>>(h1v, 512, l2_w1, l2_b1, bbv, 512, 1, 1);
  ie_tg_axpy<<<(8 * 512 + 255) / 256, 256, 0, stream>>>(bbv, 512, l2_w2, l2_b2, h1v, h2v, 512);
  ie_tg_act <<<(8 * 512 + 255) / 256, 256, 0, stream>>>(h2v, 512, lo_w, lo_b, dout, 512, 0, 0);

  (void)in_sizes; (void)n_in; (void)out_size; (void)ws_size;
}
